// WindowAttention_25718264168752
// MI455X (gfx1250) — matrix operand order repaired and checked by reading, not compiled, not run on hardware
//
#include <hip/hip_runtime.h>
#include <cstdint>
#include <cstddef>

typedef __bf16 bf16_t;
typedef __attribute__((ext_vector_type(4)))  __bf16 v4bf;
typedef __attribute__((ext_vector_type(8)))  __bf16 v8bf;
typedef __attribute__((ext_vector_type(16))) __bf16 v16bf;
typedef __attribute__((ext_vector_type(8)))  float  v8f;
typedef __attribute__((__vector_size__(4 * sizeof(int)))) int i4;

#define NTOK    49
#define HEADS   12
#define HD      32
#define CDIM    384
#define QKVDIM  1152
#define ATTN_SCALE 0.17677669529663687f   // 32^-0.5
#define NEG_BIG    -3.0e38f

// CDNA5 async global->LDS copy (ASYNCcnt path), guarded so the file compiles
// on toolchains without the builtin.
#if defined(__has_builtin)
#if __has_builtin(__builtin_amdgcn_global_load_async_to_lds_b128)
#define HAVE_ASYNC_LDS 1
#endif
#endif

#if defined(HAVE_ASYNC_LDS)
__device__ __forceinline__ void async_copy16(const void* g, void* l) {
  __builtin_amdgcn_global_load_async_to_lds_b128(
      (__attribute__((address_space(1))) i4*)(uintptr_t)g,
      (__attribute__((address_space(3))) i4*)(__attribute__((address_space(3))) void*)l,
      0, 0);
}
#define ASYNC_WAIT() asm volatile("s_wait_asynccnt 0" ::: "memory")
#else
#define ASYNC_WAIT()
#endif

// ---------------- WMMA fragment loaders (CDNA5 16x16x32 bf16 layouts) -------
// A operand (16x32, M rows): lane L<16 -> row L, K pairs {0..7},{16..23};
//                            lane L+16 -> row L, K pairs {8..15},{24..31}.
__device__ __forceinline__ v16bf frag_a(const bf16_t* p0, int row, int stride) {
  const int lane = threadIdx.x & 31;
  const bf16_t* p = p0 + (row + (lane & 15)) * stride + ((lane >> 4) << 3);
  v8bf lo = *(const v8bf*)(p);
  v8bf hi = *(const v8bf*)(p + 16);
  return __builtin_shufflevector(lo, hi, 0,1,2,3,4,5,6,7,8,9,10,11,12,13,14,15);
}
// B operand (32x16, N cols): lane L<16 -> col L, K {0..7},{16..23}; lane L+16 -> col L, K {8..15},{24..31}.
__device__ __forceinline__ v16bf frag_b(const bf16_t* p0, int col, int stride) {
  const int lane = threadIdx.x & 31;
  const bf16_t* p = p0 + (col + (lane & 15)) * stride + ((lane >> 4) << 3);
  v8bf lo = *(const v8bf*)(p);
  v8bf hi = *(const v8bf*)(p + 16);
  return __builtin_shufflevector(lo, hi, 0,1,2,3,4,5,6,7,8,9,10,11,12,13,14,15);
}

__device__ __forceinline__ v8f wmma_bf16(v16bf a, v16bf b, v8f c) {
  return __builtin_amdgcn_wmma_f32_16x16x32_bf16(false, a, false, b, (short)0, c,
                                                 false, false);
}

// ---------------- GEMM: C(MxN) = A(Mx384) * B(Nx384)^T + bias ---------------
// Block tile 128(M) x 128(N), 256 threads = 8 wave32 waves (4 M x 2 N); each
// wave computes 32x64 (2x4 WMMA tiles) -> 8 WMMA per 12 ds_load_b128.
// K streamed as 64-wide LDS panels (36.9 KB LDS).
template<typename AT, typename OT>
__global__ __launch_bounds__(256)
void gemm_bias(const AT* __restrict__ A, const float* __restrict__ B,
               const float* __restrict__ bias, OT* __restrict__ C, int N) {
  constexpr int KP  = 64;
  constexpr int LDA = 72;                 // 144B row stride: conflict-free frags
  __shared__ bf16_t As[128 * LDA];
  __shared__ bf16_t Bs[128 * LDA];

  const int tid = threadIdx.x;
  const long m0 = (long)blockIdx.y * 128;
  const int  n0 = blockIdx.x * 128;
  const int  wave = tid >> 5;
  const int  wm = (wave & 3) * 32;        // 4 M positions
  const int  wn = (wave >> 2) * 64;       // 2 N positions

  v8f acc[2][4] = {};
  for (int kp = 0; kp < 384; kp += KP) {
    __syncthreads();
    // ---- A panel: 128 x 64 -> bf16 LDS
    if constexpr (sizeof(AT) == 4) {       // fp32 source, convert on the fly
      for (int i = tid; i < 128 * (KP / 4); i += 256) {
        int r = i >> 4, c = (i & 15) << 2;
        const float4 f = *(const float4*)((const float*)A + (m0 + r) * 384 + kp + c);
        v4bf o; o.x = (bf16_t)f.x; o.y = (bf16_t)f.y; o.z = (bf16_t)f.z; o.w = (bf16_t)f.w;
        *(v4bf*)(As + r * LDA + c) = o;
      }
    } else {                               // bf16 source: straight copy
      for (int i = tid; i < 128 * (KP / 8); i += 256) {
        int r = i >> 3, c = (i & 7) << 3;
#if defined(HAVE_ASYNC_LDS)
        async_copy16((const bf16_t*)A + (m0 + r) * 384 + kp + c, As + r * LDA + c);
#else
        *(v8bf*)(As + r * LDA + c) =
            *(const v8bf*)((const bf16_t*)A + (m0 + r) * 384 + kp + c);
#endif
      }
    }
    // ---- B panel: 128 x 64 fp32 weights -> bf16 LDS
    for (int i = tid; i < 128 * (KP / 4); i += 256) {
      int r = i >> 4, c = (i & 15) << 2;
      const float4 f = *(const float4*)(B + (long)(n0 + r) * 384 + kp + c);
      v4bf o; o.x = (bf16_t)f.x; o.y = (bf16_t)f.y; o.z = (bf16_t)f.z; o.w = (bf16_t)f.w;
      *(v4bf*)(Bs + r * LDA + c) = o;
    }
    ASYNC_WAIT();
    __syncthreads();
    // ---- 2 WMMA K-steps over this panel
    #pragma unroll
    for (int kk = 0; kk < KP; kk += 32) {
      v16bf a0 = frag_a(As + kk, wm,      LDA);
      v16bf a1 = frag_a(As + kk, wm + 16, LDA);
      v16bf b0 = frag_b(Bs + kk, wn,      LDA);
      v16bf b1 = frag_b(Bs + kk, wn + 16, LDA);
      v16bf b2 = frag_b(Bs + kk, wn + 32, LDA);
      v16bf b3 = frag_b(Bs + kk, wn + 48, LDA);
      acc[0][0] = wmma_bf16(a0, b0, acc[0][0]);
      acc[0][1] = wmma_bf16(a0, b1, acc[0][1]);
      acc[0][2] = wmma_bf16(a0, b2, acc[0][2]);
      acc[0][3] = wmma_bf16(a0, b3, acc[0][3]);
      acc[1][0] = wmma_bf16(a1, b0, acc[1][0]);
      acc[1][1] = wmma_bf16(a1, b1, acc[1][1]);
      acc[1][2] = wmma_bf16(a1, b2, acc[1][2]);
      acc[1][3] = wmma_bf16(a1, b3, acc[1][3]);
    }
  }
  // ---- epilogue: bias add + store (C layout: lane halves = N cols, vgpr = M row)
  const int lane = tid & 31;
  const int cl = lane & 15, rh = (lane >> 4) * 8;
  #pragma unroll
  for (int mt = 0; mt < 2; mt++) {
    #pragma unroll
    for (int nt = 0; nt < 4; nt++) {
      const int  col  = n0 + wn + nt * 16 + cl;
      const float bv  = bias[col];
      const long rowb = m0 + wm + mt * 16 + rh;
      #pragma unroll
      for (int v = 0; v < 8; v++)
        C[(rowb + v) * N + col] = (OT)(acc[mt][nt][v] + bv);
    }
  }
}

// ---------------- Fused window attention: one block per (window, head) ------
__global__ __launch_bounds__(128)
void win_attn(const bf16_t* __restrict__ qkv, const float* __restrict__ mask,
              const float* __restrict__ bias_table, bf16_t* __restrict__ aout) {
  __shared__ bf16_t qs[64 * 40];       // q tile, rows padded to 64 (zeros)
  __shared__ bf16_t ks[64 * 40];       // k tile (B operand of QK^T)
  __shared__ bf16_t vts[32 * 72];      // v^T  (B operand of AV), keys padded
  __shared__ bf16_t ats[64 * 72];      // softmaxed probs (A operand of AV)
  __shared__ bf16_t vstage[NTOK * HD];
  __shared__ float  biash[169];        // bias_table[:, h]
  __shared__ float  bm[64 * 64];       // bias+mask, padded entries = NEG_BIG

  const int tid = threadIdx.x;
  const int b = blockIdx.y, h = blockIdx.x;
  const int w = b & 63;
  const long base = (long)b * NTOK * QKVDIM + (long)h * HD;

  // phase 1: zero padded operand tiles, stage per-head bias column
  for (int i = tid; i < 64 * 40 / 2; i += 128) ((uint32_t*)qs)[i]  = 0u;
  for (int i = tid; i < 64 * 40 / 2; i += 128) ((uint32_t*)ks)[i]  = 0u;
  for (int i = tid; i < 32 * 72 / 2; i += 128) ((uint32_t*)vts)[i] = 0u;
  for (int i = tid; i < 169; i += 128) biash[i] = bias_table[i * HEADS + h];
  __syncthreads();

  // phase 2: q/k/v tile loads (async global->LDS when available) and the
  // padded bias+mask tile: valid (i,j) -> bias+mask, padding -> NEG_BIG.
  for (int i = tid; i < NTOK * 4; i += 128) {
    int r = i >> 2, c = (i & 3) << 3;
    const bf16_t* src = qkv + base + (long)r * QKVDIM + c;
#if defined(HAVE_ASYNC_LDS)
    async_copy16(src,       qs + r * 40 + c);
    async_copy16(src + 384, ks + r * 40 + c);
    async_copy16(src + 768, vstage + r * HD + c);
#else
    *(v8bf*)(qs + r * 40 + c)     = *(const v8bf*)(src);
    *(v8bf*)(ks + r * 40 + c)     = *(const v8bf*)(src + 384);
    *(v8bf*)(vstage + r * HD + c) = *(const v8bf*)(src + 768);
#endif
  }
  for (int idx = tid; idx < 64 * 64; idx += 128) {
    const int i = idx >> 6, j = idx & 63;
    float v = NEG_BIG;
    if (i < NTOK && j < NTOK) {
      const int yi = i / 7, xi = i - yi * 7;
      const int yj = j / 7, xj = j - yj * 7;
      v = biash[(yi - yj + 6) * 13 + (xi - xj + 6)] +
          mask[(long)w * NTOK * NTOK + i * NTOK + j];
    }
    bm[idx] = v;
  }
  ASYNC_WAIT();
  __syncthreads();

  // phase 3: transpose v into vts[d][key]
  for (int i = tid; i < NTOK * HD; i += 128) {
    int key = i >> 5, d = i & 31;
    vts[d * 72 + key] = vstage[i];
  }
  __syncthreads();

  const int wv = tid >> 5, lane = tid & 31;
  const int mb = wv * 16;                    // this wave's 16 query rows
  const int cl = lane & 15, rh = (lane >> 4) * 8;

  // ---- S = Q K^T  (K-dim = hd = 32: single WMMA step, 4 N-tiles)
  v8f s[4] = {};
  {
    v16bf a = frag_a(qs, mb, 40);
    #pragma unroll
    for (int t = 0; t < 4; t++) {
      v16bf bb = frag_b(ks, t * 16, 40);
      s[t] = wmma_bf16(a, bb, s[t]);
    }
  }

  // ---- logits = S*scale + padded(bias+mask): branch-free, no exec masking
  float lg[4][8];
  #pragma unroll
  for (int t = 0; t < 4; t++) {
    const int j = t * 16 + cl;
    #pragma unroll
    for (int v = 0; v < 8; v++) {
      const int i = mb + rh + v;
      lg[t][v] = fmaf(s[t][v], ATTN_SCALE, bm[(i << 6) + j]);
    }
  }

  // ---- row softmax across the 16 lanes holding each row (xor 1..8)
  float sm[8];
  #pragma unroll
  for (int v = 0; v < 8; v++) {
    float m = fmaxf(fmaxf(lg[0][v], lg[1][v]), fmaxf(lg[2][v], lg[3][v]));
    #pragma unroll
    for (int off = 1; off <= 8; off <<= 1) m = fmaxf(m, __shfl_xor(m, off, 32));
    float t0 = __expf(lg[0][v] - m), t1 = __expf(lg[1][v] - m);
    float t2 = __expf(lg[2][v] - m), t3 = __expf(lg[3][v] - m);
    lg[0][v] = t0; lg[1][v] = t1; lg[2][v] = t2; lg[3][v] = t3;
    float ss = t0 + t1 + t2 + t3;
    #pragma unroll
    for (int off = 1; off <= 8; off <<= 1) ss += __shfl_xor(ss, off, 32);
    sm[v] = 1.0f / ss;
  }
  // probs -> bf16 A-operand tile in LDS (wave-private rows; DS ops in order)
  #pragma unroll
  for (int t = 0; t < 4; t++)
    #pragma unroll
    for (int v = 0; v < 8; v++)
      ats[(mb + rh + v) * 72 + (t * 16 + cl)] = (bf16_t)(lg[t][v] * sm[v]);

  // ---- O = P V  (K-dim = 64 keys: 2 WMMA K-steps, 2 N-tiles of hd)
  v8f o[2] = {};
  #pragma unroll
  for (int kk = 0; kk < 2; kk++) {
    v16bf a = frag_a(ats + kk * 32, mb, 72);
    #pragma unroll
    for (int nt = 0; nt < 2; nt++) {
      v16bf bb = frag_b(vts + kk * 32, nt * 16, 72);
      o[nt] = wmma_bf16(a, bb, o[nt]);
    }
  }
  const long ob = (long)b * NTOK * CDIM + (long)h * HD;
  #pragma unroll
  for (int nt = 0; nt < 2; nt++)
    #pragma unroll
    for (int v = 0; v < 8; v++) {
      const int i = mb + rh + v;
      if (i < NTOK) aout[ob + (long)i * CDIM + nt * 16 + cl] = (bf16_t)o[nt][v];
    }
}

// ---------------- launcher ---------------------------------------------------
extern "C" void kernel_launch(void* const* d_in, const int* in_sizes, int n_in,
                              void* d_out, int out_size, void* d_ws, size_t ws_size,
                              hipStream_t stream) {
  (void)in_sizes; (void)n_in; (void)out_size; (void)ws_size;
  const float* x          = (const float*)d_in[0];
  const float* mask       = (const float*)d_in[1];
  const float* qkv_w      = (const float*)d_in[2];
  const float* qkv_b      = (const float*)d_in[3];
  const float* proj_w     = (const float*)d_in[4];
  const float* proj_b     = (const float*)d_in[5];
  const float* bias_table = (const float*)d_in[6];

  const int  BW = 4096;
  const long M  = (long)BW * NTOK;              // 200704 rows
  bf16_t* qkv  = (bf16_t*)d_ws;                 // [M, 1152] bf16
  bf16_t* aout = qkv + M * QKVDIM;              // [M, 384]  bf16

  // 1) QKV projection: x(fp32) @ qkv_w^T + qkv_b -> bf16
  gemm_bias<float, bf16_t><<<dim3(QKVDIM / 128, (unsigned)(M / 128)), dim3(256), 0, stream>>>(
      x, qkv_w, qkv_b, qkv, QKVDIM);
  // 2) fused per-(window,head) attention with bias/mask/softmax
  win_attn<<<dim3(HEADS, BW), dim3(128), 0, stream>>>(qkv, mask, bias_table, aout);
  // 3) output projection: aout(bf16) @ proj_w^T + proj_b -> fp32 d_out
  gemm_bias<bf16_t, float><<<dim3(CDIM / 128, (unsigned)(M / 128)), dim3(256), 0, stream>>>(
      aout, proj_w, proj_b, (float*)d_out, CDIM);
}
